// MEGNet_62689342653100
// MI455X (gfx1250) — matrix operand order repaired and checked by reading, not compiled, not run on hardware
//
#include <hip/hip_runtime.h>
#include <stdint.h>

// ---------------------------------------------------------------------------
// MEGNet forward on gfx1250 (MI455X).
// bf16 activations + v_wmma_f32_16x16x32_bf16 (f32 accum). All GEMM operands
// padded to K%32==0 so tiles stage via global_load_async_to_lds_b128 (ASYNCcnt)
// with zero bounds checks; weights pre-converted to bf16 [64][Kpad] transposed.
// Memory-bound: ~1.5-2 GB total HBM traffic in bf16 -> ~80us @ 23.3 TB/s.
// ---------------------------------------------------------------------------

typedef uint16_t bf16;                                            // storage
typedef __attribute__((ext_vector_type(16))) __bf16 v16bf;        // WMMA A/B
typedef __attribute__((ext_vector_type(8)))  float  v8f;          // WMMA C/D

#define NNODES  25000
#define NEDGES  250000
#define NGRAPH  256
#define BN_ISTD 0.9999950000374997f

__device__ __forceinline__ float b2f(bf16 b) {
  union { uint32_t u; float f; } o; o.u = ((uint32_t)b) << 16; return o.f;
}
__device__ __forceinline__ bf16 f2b(float f) {
  union { float f; uint32_t u; } c; c.f = f;
  return (bf16)((c.u + 0x7FFFu + ((c.u >> 16) & 1u)) >> 16);      // RNE
}
__device__ __forceinline__ uint32_t pack2(float lo, float hi) {
  return (uint32_t)f2b(lo) | ((uint32_t)f2b(hi) << 16);
}

// CDNA5 async global->LDS copy (ASYNCcnt-tracked, 16B granule).
__device__ __forceinline__ void g2l_async_b128(void* lds, const void* g) {
  asm volatile("global_load_async_to_lds_b128 %0, %1, off"
               :: "v"((uint32_t)(uintptr_t)lds), "v"((uint64_t)(uintptr_t)g)
               : "memory");
}
__device__ __forceinline__ void wait_async0() {
  asm volatile("s_wait_asynccnt 0" ::: "memory");
}

// ------------------------------- helpers -----------------------------------

// f32 [rows x sc] -> bf16 [rows x dc], zero-padding cols >= sc.
__global__ void k_cvt_pad(const float* __restrict__ s, bf16* __restrict__ d,
                          int rows, int sc, int dc) {
  int t = blockIdx.x * blockDim.x + threadIdx.x;
  if (t < rows * dc) {
    int r = t / dc, c = t % dc;
    d[t] = (c < sc) ? f2b(s[(size_t)r * sc + c]) : (bf16)0;
  }
}

// W f32 [K x 64] row-major -> Wt bf16 [64 x lda] (transposed, zero-padded).
__global__ void k_cvt_w(const float* __restrict__ W, bf16* __restrict__ Wt,
                        int K, int lda) {
  int t = blockIdx.x * blockDim.x + threadIdx.x;
  if (t < 64 * lda) {
    int n = t / lda, k = t % lda;
    Wt[t] = (k < K) ? f2b(W[(size_t)k * 64 + n]) : (bf16)0;
  }
}

__global__ void k_count(const int* __restrict__ idx, float* __restrict__ cnt, int n) {
  int t = blockIdx.x * blockDim.x + threadIdx.x;
  if (t < n) atomicAdd(&cnt[idx[t]], 1.0f);
}

__global__ void k_seg_accum_bf16(const bf16* __restrict__ src, const int* __restrict__ idx,
                                 float* __restrict__ acc, int rows) {
  int t = blockIdx.x * blockDim.x + threadIdx.x;
  if (t < rows * 64) {
    int r = t >> 6, d = t & 63;
    atomicAdd(&acc[(size_t)idx[r] * 64 + d], b2f(src[t]));
  }
}

__global__ void k_seg_accum_f32(const float* __restrict__ src, const int* __restrict__ idx,
                                float* __restrict__ acc, int rows) {
  int t = blockIdx.x * blockDim.x + threadIdx.x;
  if (t < rows * 64) {
    int r = t >> 6, d = t & 63;
    atomicAdd(&acc[(size_t)idx[r] * 64 + d], src[t]);
  }
}

__global__ void k_seg_fin(const float* __restrict__ acc, const float* __restrict__ cnt,
                          int rows, bf16* __restrict__ ob, float* __restrict__ of) {
  int t = blockIdx.x * blockDim.x + threadIdx.x;
  if (t < rows * 64) {
    int r = t >> 6;
    float m = acc[t] / fmaxf(cnt[r], 1.0f);
    ob[t] = f2b(m);
    if (of) of[t] = m;
  }
}

// d = a + b elementwise over 4 bf16 per thread (n4 = n/4). In-place safe.
__global__ void k_add(bf16* __restrict__ d, const bf16* __restrict__ a,
                      const bf16* __restrict__ b, int n4) {
  int t = blockIdx.x * blockDim.x + threadIdx.x;
  if (t < n4) {
    uint2 av = ((const uint2*)a)[t], bv = ((const uint2*)b)[t];
    uint2 o;
    o.x = pack2(b2f((bf16)(av.x & 0xFFFFu)) + b2f((bf16)(bv.x & 0xFFFFu)),
                b2f((bf16)(av.x >> 16))     + b2f((bf16)(bv.x >> 16)));
    o.y = pack2(b2f((bf16)(av.y & 0xFFFFu)) + b2f((bf16)(bv.y & 0xFFFFu)),
                b2f((bf16)(av.y >> 16))     + b2f((bf16)(bv.y >> 16)));
    ((uint2*)d)[t] = o;
  }
}

// comb_e[e] = [x_t[row[e]] | x_t[col[e]] | e_t[e] | u_t[batch[row[e]]]] (256 cols)
__global__ void k_concat_edge(bf16* __restrict__ dst, const bf16* __restrict__ xt,
                              const bf16* __restrict__ et, const bf16* __restrict__ ut,
                              const int* __restrict__ row, const int* __restrict__ col,
                              const int* __restrict__ batch) {
  int t = blockIdx.x * blockDim.x + threadIdx.x;   // one uint4 (8 bf16) per thread
  if (t >= NEDGES * 32) return;
  int e = t >> 5, q = t & 31;
  int chunk = q >> 3, qi = q & 7;
  const bf16* src;
  if (chunk == 0)      src = xt + (size_t)row[e] * 64;
  else if (chunk == 1) src = xt + (size_t)col[e] * 64;
  else if (chunk == 2) src = et + (size_t)e * 64;
  else                 src = ut + (size_t)batch[row[e]] * 64;
  *(uint4*)(dst + (size_t)e * 256 + chunk * 64 + qi * 8) = *(const uint4*)(src + qi * 8);
}

// comb_n[v] = [x_t[v] | v_e[v] | u_t[batch[v]]] (192 cols)
__global__ void k_concat_node(bf16* __restrict__ dst, const bf16* __restrict__ xt,
                              const bf16* __restrict__ ve, const bf16* __restrict__ ut,
                              const int* __restrict__ batch) {
  int t = blockIdx.x * blockDim.x + threadIdx.x;
  if (t >= NNODES * 24) return;
  int v = t / 24, q = t % 24;
  int chunk = q >> 3, qi = q & 7;
  const bf16* src;
  if (chunk == 0)      src = xt + (size_t)v * 64;
  else if (chunk == 1) src = ve + (size_t)v * 64;
  else                 src = ut + (size_t)batch[v] * 64;
  *(uint4*)(dst + (size_t)v * 192 + chunk * 64 + qi * 8) = *(const uint4*)(src + qi * 8);
}

// dst[r] = [a[r] | b[r] | c[r]] (3x64 -> 192 cols)
__global__ void k_concat3(bf16* __restrict__ dst, const bf16* __restrict__ a,
                          const bf16* __restrict__ b, const bf16* __restrict__ c, int rows) {
  int t = blockIdx.x * blockDim.x + threadIdx.x;
  if (t >= rows * 24) return;
  int r = t / 24, q = t % 24;
  int chunk = q >> 3, qi = q & 7;
  const bf16* src = (chunk == 0) ? a : (chunk == 1) ? b : c;
  *(uint4*)(dst + (size_t)r * 192 + chunk * 64 + qi * 8) =
      *(const uint4*)(src + (size_t)r * 64 + qi * 8);
}

// ------------------------ WMMA GEMM: out = act(A@W + b) ---------------------
// A: bf16 [M x lda] row-major, lda%32==0, cols >= logical K zero-padded.
// Wt: bf16 [64 x lda] (weight transposed + padded). Output bf16 [M x 64].
// Block: 256 thr = 8 waves; wave owns 32x64 (2 m-tiles x 4 n-tiles, 8 WMMA/Kstep).
// Fast path (full 256-row block): async b128 global->LDS staging, no guards.

__launch_bounds__(256)
__global__ void k_wmma_gemm64(const bf16* __restrict__ A, int M, int lda,
                              const bf16* __restrict__ Wt,
                              const float* __restrict__ Bi,
                              const float* __restrict__ G, const float* __restrict__ Be,
                              bf16* __restrict__ out, int do_relu) {
  __shared__ bf16 sA[256][40];   // 256 rows x 32 K, stride 40 (16B-aligned rows)
  __shared__ bf16 sW[64][40];    // Wt tile: [n][k]

  const int tid = threadIdx.x, lane = tid & 31, w = tid >> 5;
  const int blockRow = blockIdx.x * 256;
  const bool full = (blockRow + 256 <= M);

  v8f acc[2][4];
#pragma unroll
  for (int i = 0; i < 2; ++i)
#pragma unroll
    for (int j = 0; j < 4; ++j)
      acc[i][j] = (v8f){0.f, 0.f, 0.f, 0.f, 0.f, 0.f, 0.f, 0.f};

  union Frag { v16bf v; uint32_t u[8]; };

  for (int k0 = 0; k0 < lda; k0 += 32) {
    if (full) {
#pragma unroll
      for (int it = 0; it < 4; ++it) {              // 1024 b128 chunks / 256 thr
        int q = tid + it * 256;
        int r = q >> 2, c8 = (q & 3) << 3;
        g2l_async_b128(&sA[r][c8], A + (size_t)(blockRow + r) * lda + k0 + c8);
      }
    } else {                                        // M-tail block only
      for (int i = tid; i < 256 * 32; i += 256) {
        int r = i >> 5, c = i & 31, gr = blockRow + r;
        sA[r][c] = (gr < M) ? A[(size_t)gr * lda + k0 + c] : (bf16)0;
      }
    }
    { int n = tid >> 2, c8 = (tid & 3) << 3;        // 256 b128 chunks / 256 thr
      g2l_async_b128(&sW[n][c8], Wt + (size_t)n * lda + k0 + c8); }
    wait_async0();
    __syncthreads();

    // B fragments (32x16 bf16): lanes 0-15 K{0-7,16-23}, lanes 16-31 K{8-15,24-31}.
    const int nn = lane & 15;
    const int kbB = (lane < 16) ? 0 : 8;
    Frag bfr[4];
#pragma unroll
    for (int nt = 0; nt < 4; ++nt)
#pragma unroll
      for (int j = 0; j < 8; ++j)
        bfr[nt].u[j] = *reinterpret_cast<const uint32_t*>(&sW[nt * 16 + nn][((j < 4) ? (2 * j) : (8 + 2 * j)) + kbB]);

    // A fragments (16x32): lanes 0-15 K{0-7,16-23}, lanes 16-31 K{8-15,24-31}.
    const int kbA = (lane < 16) ? 0 : 8;
#pragma unroll
    for (int mt = 0; mt < 2; ++mt) {
      const int m = w * 32 + mt * 16 + (lane & 15);
      Frag af;
#pragma unroll
      for (int j = 0; j < 8; ++j) {
        int koff = ((j < 4) ? (2 * j) : (8 + 2 * j)) + kbA;   // j>=4 -> 16+2*(j-4)
        af.u[j] = *reinterpret_cast<const uint32_t*>(&sA[m][koff]);
      }
#pragma unroll
      for (int nt = 0; nt < 4; ++nt)
        acc[mt][nt] = __builtin_amdgcn_wmma_f32_16x16x32_bf16(
            false, af.v, false, bfr[nt].v, (short)0, acc[mt][nt], false, false);
    }
    __syncthreads();
  }

  // Epilogue: bias + relu (+ eval BN). C layout: VGPR v -> M=v+8*(lane>=16), N=lane%16.
#pragma unroll
  for (int mt = 0; mt < 2; ++mt)
#pragma unroll
    for (int nt = 0; nt < 4; ++nt) {
      int ncol = nt * 16 + (lane & 15);
      float bia = Bi[ncol];
      float g = 1.f, be = 0.f;
      if (G) { g = G[ncol]; be = Be[ncol]; }
#pragma unroll
      for (int v = 0; v < 8; ++v) {
        int mrow = blockRow + w * 32 + mt * 16 + v + ((lane < 16) ? 0 : 8);
        if (mrow < M) {
          float h = acc[mt][nt][v] + bia;
          if (do_relu) h = fmaxf(h, 0.f);
          if (G) h = g * (h * BN_ISTD) + be;
          out[(size_t)mrow * 64 + ncol] = f2b(h);
        }
      }
    }
}

// Final 64->1 linear (no activation), writes f32 output.
__global__ void k_linout(const bf16* __restrict__ h, const float* __restrict__ Wo,
                         const float* __restrict__ bo, float* __restrict__ out) {
  int t = blockIdx.x * blockDim.x + threadIdx.x;
  if (t < NGRAPH) {
    float s = bo[0];
#pragma unroll
    for (int k = 0; k < 64; ++k) s += b2f(h[t * 64 + k]) * Wo[k];
    out[t] = s;
  }
}

// ------------------------------- host side ----------------------------------

static inline int cdiv(int a, int b) { return (a + b - 1) / b; }
static inline int rup32(int k) { return (k + 31) & ~31; }

extern "C" void kernel_launch(void* const* d_in, const int* in_sizes, int n_in,
                              void* d_out, int out_size, void* d_ws, size_t ws_size,
                              hipStream_t stream) {
  (void)in_sizes; (void)n_in; (void)out_size; (void)ws_size;
  const int B = 256;

  // Inputs in setup_inputs() insertion order; params in jax pytree order
  // (dict keys sorted): gc[0..2]{e_embed,edge{bn,lin},global{bn,lin},
  // node{bn,lin},u_embed,x_embed} (48 leaves each), lin_out, post, pre.
  const float* X     = (const float*)d_in[0];
  const float* EA    = (const float*)d_in[1];
  const int*   EI    = (const int*)d_in[2];
  const float* U     = (const float*)d_in[3];
  const int*   BATCH = (const int*)d_in[4];
  const int*   ROW   = EI;
  const int*   COL   = EI + NEDGES;
  auto P = [&](int i) { return (const float*)d_in[5 + i]; };

  // Workspace bump allocator (256B aligned).
  uintptr_t wsb = (uintptr_t)d_ws;
  size_t off = 0;
  auto alloc = [&](size_t bytes) -> void* {
    void* p = (void*)(wsb + off);
    off += (bytes + 255) & ~(size_t)255;
    return p;
  };
  bf16* x_bf   = (bf16*)alloc((size_t)NNODES * 128 * 2);   // 100 -> 128 pad
  bf16* ea_bf  = (bf16*)alloc((size_t)NEDGES * 64 * 2);    // 50 -> 64 pad
  bf16* u_bf   = (bf16*)alloc((size_t)NGRAPH * 32 * 2);    // 3 -> 32 pad
  bf16* Wt     = (bf16*)alloc((size_t)64 * 256 * 2);       // weight scratch
  bf16* xc     = (bf16*)alloc((size_t)NNODES * 64 * 2);
  bf16* ec     = (bf16*)alloc((size_t)NEDGES * 64 * 2);
  bf16* uc     = (bf16*)alloc((size_t)NGRAPH * 64 * 2);
  bf16* x_t    = (bf16*)alloc((size_t)NNODES * 64 * 2);
  bf16* e_t    = (bf16*)alloc((size_t)NEDGES * 64 * 2);
  bf16* u_t    = (bf16*)alloc((size_t)NGRAPH * 64 * 2);
  bf16* tE1    = (bf16*)alloc((size_t)NEDGES * 64 * 2);
  bf16* tE2    = (bf16*)alloc((size_t)NEDGES * 64 * 2);
  bf16* tN1    = (bf16*)alloc((size_t)NNODES * 64 * 2);
  bf16* tN2    = (bf16*)alloc((size_t)NNODES * 64 * 2);
  bf16* tU1    = (bf16*)alloc((size_t)NGRAPH * 64 * 2);
  bf16* tU2    = (bf16*)alloc((size_t)NGRAPH * 64 * 2);
  bf16* comb_e = (bf16*)alloc((size_t)NEDGES * 256 * 2);
  bf16* comb_n = (bf16*)alloc((size_t)NNODES * 192 * 2);
  bf16* comb_u = (bf16*)alloc((size_t)NGRAPH * 192 * 2);
  bf16* e_out  = (bf16*)alloc((size_t)NEDGES * 64 * 2);
  bf16* x_out  = (bf16*)alloc((size_t)NNODES * 64 * 2);
  bf16* u_out  = (bf16*)alloc((size_t)NGRAPH * 64 * 2);
  float* node_acc  = (float*)alloc((size_t)NNODES * 64 * 4);
  float* node_mean = (float*)alloc((size_t)NNODES * 64 * 4);
  bf16* v_e    = (bf16*)alloc((size_t)NNODES * 64 * 2);
  float* g_acc = (float*)alloc((size_t)NGRAPH * 64 * 4);
  bf16* u_e    = (bf16*)alloc((size_t)NGRAPH * 64 * 2);
  bf16* u_v    = (bf16*)alloc((size_t)NGRAPH * 64 * 2);
  bf16* x_pool = (bf16*)alloc((size_t)NGRAPH * 64 * 2);
  bf16* e_pool = (bf16*)alloc((size_t)NGRAPH * 64 * 2);
  float* cnt_node  = (float*)alloc((size_t)NNODES * 4);
  float* cnt_graph = (float*)alloc((size_t)NGRAPH * 4);
  bf16* post_h = (bf16*)alloc((size_t)NGRAPH * 64 * 2);

  // GEMM = weight convert (bf16, transposed, padded) + WMMA kernel.
  auto gemm = [&](const bf16* A, int M, int K, int wi, int bi, int gi, int bei,
                  bf16* outp, int relu) {
    int lda = rup32(K);
    k_cvt_w<<<cdiv(64 * lda, B), B, 0, stream>>>(P(wi), Wt, K, lda);
    k_wmma_gemm64<<<cdiv(M, 256), 256, 0, stream>>>(
        A, M, lda, Wt, P(bi), gi >= 0 ? P(gi) : nullptr,
        bei >= 0 ? P(bei) : nullptr, outp, relu);
  };

  // Input conversions (padded) + counts (row-degree, graph node count).
  k_cvt_pad<<<cdiv(NNODES * 128, B), B, 0, stream>>>(X, x_bf, NNODES, 100, 128);
  k_cvt_pad<<<cdiv(NEDGES * 64, B), B, 0, stream>>>(EA, ea_bf, NEDGES, 50, 64);
  k_cvt_pad<<<cdiv(NGRAPH * 32, B), B, 0, stream>>>(U, u_bf, NGRAPH, 3, 32);
  hipMemsetAsync(cnt_node, 0, (size_t)NNODES * 4, stream);
  hipMemsetAsync(cnt_graph, 0, (size_t)NGRAPH * 4, stream);
  k_count<<<cdiv(NEDGES, B), B, 0, stream>>>(ROW, cnt_node, NEDGES);
  k_count<<<cdiv(NNODES, B), B, 0, stream>>>(BATCH, cnt_graph, NNODES);

  // pre: xc = relu(x @ Wpre + bpre)
  gemm(x_bf, NNODES, 100, 148, 149, -1, -1, xc, 1);

  const bf16* ecur = ea_bf; int Ke = 50;
  const bf16* ucur = u_bf;  int Ku = 3;

  for (int L = 0; L < 3; ++L) {
    int b = 48 * L;
    // embeds (Linear+ReLU x2)
    gemm(ecur, NEDGES, Ke, b + 0, b + 1, -1, -1, tE1, 1);
    gemm(tE1, NEDGES, 64, b + 2, b + 3, -1, -1, e_t, 1);
    gemm(xc, NNODES, 64, b + 44, b + 45, -1, -1, tN1, 1);
    gemm(tN1, NNODES, 64, b + 46, b + 47, -1, -1, x_t, 1);
    gemm(ucur, NGRAPH, Ku, b + 40, b + 41, -1, -1, tU1, 1);
    gemm(tU1, NGRAPH, 64, b + 42, b + 43, -1, -1, u_t, 1);
    // edge model
    k_concat_edge<<<cdiv(NEDGES * 32, B), B, 0, stream>>>(comb_e, x_t, e_t, u_t,
                                                          ROW, COL, BATCH);
    gemm(comb_e, NEDGES, 256, b + 10, b + 11, b + 4, b + 5, tE1, 1);
    gemm(tE1, NEDGES, 64, b + 12, b + 13, b + 6, b + 7, tE2, 1);
    gemm(tE2, NEDGES, 64, b + 14, b + 15, b + 8, b + 9, e_out, 1);
    // v_e = seg_mean(e_out, row); u_e = seg_mean(v_e, batch)
    hipMemsetAsync(node_acc, 0, (size_t)NNODES * 64 * 4, stream);
    k_seg_accum_bf16<<<cdiv(NEDGES * 64, B), B, 0, stream>>>(e_out, ROW, node_acc, NEDGES);
    k_seg_fin<<<cdiv(NNODES * 64, B), B, 0, stream>>>(node_acc, cnt_node, NNODES, v_e, node_mean);
    hipMemsetAsync(g_acc, 0, (size_t)NGRAPH * 64 * 4, stream);
    k_seg_accum_f32<<<cdiv(NNODES * 64, B), B, 0, stream>>>(node_mean, BATCH, g_acc, NNODES);
    k_seg_fin<<<cdiv(NGRAPH * 64, B), B, 0, stream>>>(g_acc, cnt_graph, NGRAPH, u_e, nullptr);
    // node model
    k_concat_node<<<cdiv(NNODES * 24, B), B, 0, stream>>>(comb_n, x_t, v_e, u_t, BATCH);
    gemm(comb_n, NNODES, 192, b + 34, b + 35, b + 28, b + 29, tN1, 1);
    gemm(tN1, NNODES, 64, b + 36, b + 37, b + 30, b + 31, tN2, 1);
    gemm(tN2, NNODES, 64, b + 38, b + 39, b + 32, b + 33, x_out, 1);
    // u_v = seg_mean(x_out, batch)
    hipMemsetAsync(g_acc, 0, (size_t)NGRAPH * 64 * 4, stream);
    k_seg_accum_bf16<<<cdiv(NNODES * 64, B), B, 0, stream>>>(x_out, BATCH, g_acc, NNODES);
    k_seg_fin<<<cdiv(NGRAPH * 64, B), B, 0, stream>>>(g_acc, cnt_graph, NGRAPH, u_v, nullptr);
    // global model
    k_concat3<<<cdiv(NGRAPH * 24, B), B, 0, stream>>>(comb_u, u_e, u_v, u_t, NGRAPH);
    gemm(comb_u, NGRAPH, 192, b + 22, b + 23, b + 16, b + 17, tU1, 1);
    gemm(tU1, NGRAPH, 64, b + 24, b + 25, b + 18, b + 19, tU2, 1);
    gemm(tU2, NGRAPH, 64, b + 26, b + 27, b + 20, b + 21, u_out, 1);
    // residuals (layer0 adds the embeds; later layers add carried state)
    k_add<<<cdiv(NEDGES * 16, B), B, 0, stream>>>(ec, e_out, (L == 0) ? e_t : ec, NEDGES * 16);
    k_add<<<cdiv(NNODES * 16, B), B, 0, stream>>>(xc, x_out, (L == 0) ? x_t : xc, NNODES * 16);
    k_add<<<cdiv(NGRAPH * 16, B), B, 0, stream>>>(uc, u_out, (L == 0) ? u_t : uc, NGRAPH * 16);
    ecur = ec; Ke = 64; ucur = uc; Ku = 64;
  }

  // Final pooling: x_pool, e_pool (double mean), then post MLP + lin_out.
  hipMemsetAsync(g_acc, 0, (size_t)NGRAPH * 64 * 4, stream);
  k_seg_accum_bf16<<<cdiv(NNODES * 64, B), B, 0, stream>>>(xc, BATCH, g_acc, NNODES);
  k_seg_fin<<<cdiv(NGRAPH * 64, B), B, 0, stream>>>(g_acc, cnt_graph, NGRAPH, x_pool, nullptr);

  hipMemsetAsync(node_acc, 0, (size_t)NNODES * 64 * 4, stream);
  k_seg_accum_bf16<<<cdiv(NEDGES * 64, B), B, 0, stream>>>(ec, ROW, node_acc, NEDGES);
  k_seg_fin<<<cdiv(NNODES * 64, B), B, 0, stream>>>(node_acc, cnt_node, NNODES, v_e, node_mean);
  hipMemsetAsync(g_acc, 0, (size_t)NGRAPH * 64 * 4, stream);
  k_seg_accum_f32<<<cdiv(NNODES * 64, B), B, 0, stream>>>(node_mean, BATCH, g_acc, NNODES);
  k_seg_fin<<<cdiv(NGRAPH * 64, B), B, 0, stream>>>(g_acc, cnt_graph, NGRAPH, e_pool, nullptr);

  k_concat3<<<cdiv(NGRAPH * 24, B), B, 0, stream>>>(comb_u, x_pool, e_pool, uc, NGRAPH);
  gemm(comb_u, NGRAPH, 192, 146, 147, -1, -1, post_h, 1);
  k_linout<<<1, 256, 0, stream>>>(post_h, P(144), P(145), (float*)d_out);
}
